// Core_Unique_47021301956915
// MI455X (gfx1250) — matrix operand order repaired and checked by reading, not compiled, not run on hardware
//
#include <hip/hip_runtime.h>
#include <cmath>

// ---------------------------------------------------------------------------
// Problem constants (match reference)
// ---------------------------------------------------------------------------
#define BB 2
#define CC 64
#define NN 2048
#define TT 64
#define BN_EPS 1e-5f

#define NODES1 16                 // nodes per block in kernel 1
#define NCH1 (NN / NODES1)        // 128 chunks per batch
#define NODES3 16                 // nodes per block in kernel 3
#define NCH3 (NN / NODES3)        // 128 chunks per batch

// ---------------------------------------------------------------------------
// WMMA types & helpers (gfx1250, wave32). v_wmma_f32_16x16x32_bf16:
//   A: 16x32 bf16 (v16bf), B: 32x16 bf16 (v16bf), C/D: 16x16 f32 (v8f)
// ---------------------------------------------------------------------------
typedef __attribute__((ext_vector_type(8)))  __bf16 bfx8;
typedef __attribute__((ext_vector_type(16))) __bf16 bfx16;
typedef __attribute__((ext_vector_type(8)))  float  fx8;
typedef __attribute__((ext_vector_type(4)))  unsigned int u32x4;
typedef __attribute__((ext_vector_type(8)))  int  i32x8;
typedef __attribute__((ext_vector_type(4)))  int  i32x4;

// A fragment (16x32, M x K) from LDS row-major [M][ld] bf16.
// ISA layout: lane<16 -> row M=lane, K {0..7} in v0..3 and {16..23} in v4..7;
//             lane>=16 -> row M=lane-16, K {8..15} and {24..31}.
__device__ __forceinline__ bfx16 ldA(const __bf16* base, int ld, int lane,
                                     int m0, int k0) {
  const int r = lane & 15, h = lane >> 4;
  const __bf16* p = base + (size_t)(m0 + r) * ld + k0 + h * 8;
  bfx8 lo = *(const bfx8*)p;          // ds_load_b128
  bfx8 hi = *(const bfx8*)(p + 16);   // ds_load_b128
  return __builtin_shufflevector(lo, hi, 0, 1, 2, 3, 4, 5, 6, 7,
                                 8, 9, 10, 11, 12, 13, 14, 15);
}

// B fragment (32x16, K x N) from LDS stored K-contiguous: [N][ld] bf16.
// ISA layout: lane<16 -> col N=lane, K {0..7} and {16..23}; lane>=16 -> col N=lane-16, K {8..15} and {24..31}.
__device__ __forceinline__ bfx16 ldB(const __bf16* base, int ld, int lane,
                                     int n0, int k0) {
  const int n = lane & 15, h = lane >> 4;
  const __bf16* p = base + (size_t)(n0 + n) * ld + k0 + h * 8;
  bfx8 lo = *(const bfx8*)p;
  bfx8 hi = *(const bfx8*)(p + 16);
  return __builtin_shufflevector(lo, hi, 0, 1, 2, 3, 4, 5, 6, 7,
                                 8, 9, 10, 11, 12, 13, 14, 15);
}

__device__ __forceinline__ fx8 wmma_bf16(bfx16 a, bfx16 b, fx8 c) {
  return __builtin_amdgcn_wmma_f32_16x16x32_bf16(
      /*neg_a=*/false, a, /*neg_b=*/false, b,
      /*c_mod=*/(short)0, c, /*reuse_a=*/false, /*reuse_b=*/false);
}

__device__ __forceinline__ float gelu_exact(float x) {
  return 0.5f * x * (1.0f + erff(x * 0.70710678118654752f));
}

// ---------------------------------------------------------------------------
// Tensor Data Mover: async 2D fp32 tile load (Global -> LDS).
// D# per CDNA5 ISA ch.8:  group0 = {flags, lds_addr, global_addr, type=2},
// group1 = {data_size=4B, tensor_dim0/1, tile_dim0/1, dim0 stride},
// groups 2/3 (and the trailing i32x8 of the 6-arg builtin) zero: 2-D tile,
// tile_dim2/3/4 = 0 => unused. Tracked with TENSORcnt.
// tile_x elements contiguous per row; tile_y rows at row_stride (elements).
// ---------------------------------------------------------------------------
__device__ __forceinline__ void tdm_load_2d_f32(const float* gsrc,
                                                unsigned lds_off, int tile_x,
                                                int tile_y,
                                                unsigned long long row_stride) {
  const unsigned long long ga = (unsigned long long)(uintptr_t)gsrc;
  u32x4 g0;
  g0.x = 1u;                                   // count=1, user descriptor
  g0.y = lds_off;                              // LDS byte offset
  g0.z = (unsigned)ga;                         // global_addr[31:0]
  g0.w = (unsigned)((ga >> 32) & 0x01ffffffu)  // global_addr[56:32]
         | (2u << 30);                         // type = 2 ("image")
  i32x8 g1;
  g1[0] = (int)(2u << 16);                                   // data_size=4B
  g1[1] = (int)(((unsigned)tile_x & 0xffffu) << 16);         // tensor_dim0 lo
  g1[2] = (int)(((unsigned)tile_y & 0xffffu) << 16);         // tensor_dim1 lo
  g1[3] = (int)(((unsigned)tile_x & 0xffffu) << 16);         // tile_dim0
  g1[4] = (int)((unsigned)tile_y & 0xffffu);                 // tile_dim1
  g1[5] = (int)(unsigned)(row_stride & 0xffffffffu);         // dim0 stride lo
  g1[6] = (int)(unsigned)((row_stride >> 32) & 0xffffu);     // dim0 stride hi
  g1[7] = 0;
  i32x4 g2 = {0, 0, 0, 0}, g3 = {0, 0, 0, 0};
  i32x8 g4 = {0, 0, 0, 0, 0, 0, 0, 0};
  __builtin_amdgcn_tensor_load_to_lds(g0, g1, g2, g3, g4, 0);
}

__device__ __forceinline__ unsigned lds_off_of(const void* p) {
  // LDS aperture: generic addr low 32 bits are the wave-relative LDS offset
  return (unsigned)(uintptr_t)p;
}

// ---------------------------------------------------------------------------
// Kernel 1: per-chunk online-softmax statistics.
// Block = (b, chunk of 16 nodes), 256 threads = 8 waves.
// Per node: u = gelu(W1 x + b1);  cm = W2 u + b2;  q = W5 x + b5.
// x tiles arrive via double-buffered TDM; waves keep running
// (max, sum exp, sum q*exp) accumulators across the node loop in registers.
// ---------------------------------------------------------------------------
__global__ __launch_bounds__(256) void k1_pool_partials(
    const float* __restrict__ xg,
    const float* __restrict__ W1, const float* __restrict__ b1,
    const float* __restrict__ W2, const float* __restrict__ b2,
    const float* __restrict__ W5, const float* __restrict__ b5,
    float* __restrict__ part) {
  extern __shared__ char smem[];
  __bf16* sW1 = (__bf16*)smem;      // 64x64
  __bf16* sW2 = sW1 + 4096;         // 64x64
  __bf16* sW5 = sW2 + 4096;         // 64x64
  __bf16* sX  = sW5 + 4096;         // [t][c] 64x64 bf16 (B-operand layout)
  __bf16* sU  = sX + 4096;          // [t][c] 64x64 bf16
  float*  sb1 = (float*)(sU + 4096);
  float*  sb2 = sb1 + 64;
  float*  sb5 = sb2 + 64;
  float*  sStg0 = sb5 + 64;         // TDM stage: [c][t] 64x64 fp32
  float*  sStg1 = sStg0 + 4096;     // TDM stage: double buffer

  const int tid = threadIdx.x, wave = tid >> 5, lane = tid & 31;
  const int h = lane >> 4, ln = lane & 15;
  const int b = blockIdx.x / NCH1, ch = blockIdx.x % NCH1;
  const int wave_u = __builtin_amdgcn_readfirstlane(wave);  // SGPR -> s_cbranch

  const float* xbase =
      xg + (size_t)b * CC * NN * TT + (size_t)(ch * NODES1) * TT;

  // Prologue: kick off TDM for node 0 of this chunk (wave 0 only; TDM
  // ignores EXEC, so the guard must be a scalar branch).
  if (wave_u == 0)
    tdm_load_2d_f32(xbase, lds_off_of(sStg0), TT, CC,
                    (unsigned long long)NN * TT);

  for (int i = tid; i < 4096; i += 256) {
    sW1[i] = (__bf16)W1[i];
    sW2[i] = (__bf16)W2[i];
    sW5[i] = (__bf16)W5[i];
  }
  if (tid < 64) { sb1[tid] = b1[tid]; sb2[tid] = b2[tid]; sb5[tid] = b5[tid]; }

  const int slot0 = wave * 2;       // 16 slots = 4 c-tiles x 4 t-tiles
  fx8 accM[2], accS[2], accW[2];
  for (int j = 0; j < 2; ++j)
    for (int v = 0; v < 8; ++v) {
      accM[j][v] = -INFINITY; accS[j][v] = 0.f; accW[j][v] = 0.f;
    }

  for (int i = 0; i < NODES1; ++i) {
    float* cur = (i & 1) ? sStg1 : sStg0;
    float* nxt = (i & 1) ? sStg0 : sStg1;
    if (wave_u == 0) __builtin_amdgcn_s_wait_tensorcnt(0);  // tile i landed
    __syncthreads();  // broadcast arrival; prior iter LDS reads complete
    if (wave_u == 0 && i + 1 < NODES1)  // prefetch next tile into other buf
      tdm_load_2d_f32(xbase + (size_t)(i + 1) * TT, lds_off_of(nxt), TT, CC,
                      (unsigned long long)NN * TT);

    // convert fp32 [c][t] stage -> bf16 [t][c] B-operand tile
    for (int idx = tid; idx < 4096; idx += 256) {
      const int c = idx >> 6, t = idx & 63;
      sX[t * 64 + c] = (__bf16)cur[c * 64 + t];
    }
    __syncthreads();

    // GEMM1: u = gelu(W1 x + b1), full 64x64, 2 fragments per wave -> LDS
#pragma unroll
    for (int j = 0; j < 2; ++j) {
      const int slot = slot0 + j, mt = slot >> 2, tt = slot & 3;
      fx8 acc = {};
#pragma unroll
      for (int k0 = 0; k0 < 64; k0 += 32)
        acc = wmma_bf16(ldA(sW1, 64, lane, mt * 16, k0),
                        ldB(sX, 64, lane, tt * 16, k0), acc);
      const int pos = tt * 16 + ln;
      bfx8 uo;
#pragma unroll
      for (int v = 0; v < 8; ++v)
        uo[v] = (__bf16)gelu_exact(acc[v] + sb1[mt * 16 + h * 8 + v]);
      *(bfx8*)(sU + pos * 64 + mt * 16 + h * 8) = uo;
    }
    __syncthreads();

    // cm = W2 u + b2 ; q = W5 x + b5 ; online softmax update
#pragma unroll
    for (int j = 0; j < 2; ++j) {
      const int slot = slot0 + j, ct = slot >> 2, tt = slot & 3;
      fx8 cm = {}, qq = {};
#pragma unroll
      for (int k0 = 0; k0 < 64; k0 += 32) {
        cm = wmma_bf16(ldA(sW2, 64, lane, ct * 16, k0),
                       ldB(sU, 64, lane, tt * 16, k0), cm);
        qq = wmma_bf16(ldA(sW5, 64, lane, ct * 16, k0),
                       ldB(sX, 64, lane, tt * 16, k0), qq);
      }
#pragma unroll
      for (int v = 0; v < 8; ++v) {
        const int row = ct * 16 + h * 8 + v;
        const float cmv = cm[v] + sb2[row];
        const float qv = qq[v] + sb5[row];
        const float om = accM[j][v];
        const float nm = fmaxf(om, cmv);
        const float sc = __expf(om - nm);
        const float p = __expf(cmv - nm);
        accS[j][v] = accS[j][v] * sc + p;
        accW[j][v] = accW[j][v] * sc + p * qv;
        accM[j][v] = nm;
      }
    }
  }

  // partials: [b][chunk][{m,s,w}][c][t]
  const size_t base = (size_t)(b * NCH1 + ch) * 3 * (CC * TT);
#pragma unroll
  for (int j = 0; j < 2; ++j) {
    const int slot = slot0 + j, ct = slot >> 2, tt = slot & 3;
    const int t = tt * 16 + ln;
#pragma unroll
    for (int v = 0; v < 8; ++v) {
      const int c = ct * 16 + h * 8 + v;
      part[base + 0 * (CC * TT) + c * TT + t] = accM[j][v];
      part[base + 1 * (CC * TT) + c * TT + t] = accS[j][v];
      part[base + 2 * (CC * TT) + c * TT + t] = accW[j][v];
    }
  }
}

// ---------------------------------------------------------------------------
// Kernel 2: merge per-chunk softmax partials -> pooled[b][c][t] = w / s
// ---------------------------------------------------------------------------
__global__ __launch_bounds__(256) void k2_pool_reduce(
    const float* __restrict__ part, float* __restrict__ pooled) {
  const int idx = blockIdx.x * 256 + threadIdx.x;  // over B*C*T = 8192
  if (idx >= BB * CC * TT) return;
  const int b = idx / (CC * TT);
  const int ct = idx % (CC * TT);
  float m = -INFINITY, s = 0.f, w = 0.f;
  for (int ch = 0; ch < NCH1; ++ch) {
    const size_t base = (size_t)(b * NCH1 + ch) * 3 * (CC * TT) + ct;
    const float mi = part[base];
    const float si = part[base + CC * TT];
    const float wi = part[base + 2 * CC * TT];
    const float nm = fmaxf(m, mi);
    const float e0 = __expf(m - nm), e1 = __expf(mi - nm);
    s = s * e0 + si * e1;
    w = w * e0 + wi * e1;
    m = nm;
  }
  pooled[idx] = w / s;
}

// ---------------------------------------------------------------------------
// Kernel 3: fused  cat=[x-pool, pool] -> gelu(W3 cat + b3) -> W4 . + b4
//           -> +x residual -> BatchNorm(eval).  Second streaming pass over x
//           (L2-resident: x = 64MB < 192MB global L2).  The TDM stage
//           buffers double as the fp32 residual tiles.
// ---------------------------------------------------------------------------
__global__ __launch_bounds__(256) void k3_mlp_bn(
    const float* __restrict__ xg, const float* __restrict__ pooled,
    const float* __restrict__ W3, const float* __restrict__ b3,
    const float* __restrict__ W4, const float* __restrict__ b4,
    const float* __restrict__ gamma, const float* __restrict__ beta,
    const float* __restrict__ rmean, const float* __restrict__ rvar,
    float* __restrict__ out) {
  extern __shared__ char smem[];
  __bf16* sW3 = (__bf16*)smem;        // 256x128 bf16 (64 KB)
  __bf16* sW4 = sW3 + 256 * 128;      // 64x256 bf16 (32 KB)
  __bf16* sCat = sW4 + 64 * 256;      // [t][128]  bf16 (16 KB)
  __bf16* sH = sCat + 64 * 128;       // [t][256]  bf16 (32 KB)
  float* sXf0 = (float*)(sH + 64 * 256);  // TDM stage A: [c][t] f32 (16 KB)
  float* sXf1 = sXf0 + 4096;              // TDM stage B: [c][t] f32 (16 KB)
  float* sPool = sXf1 + 4096;             // [c][t] f32 (16 KB)
  float* sb3 = sPool + 4096;              // 256
  float* sb4 = sb3 + 256;                 // 64
  float* sScale = sb4 + 64;               // 64
  float* sShift = sScale + 64;            // 64

  const int tid = threadIdx.x, wave = tid >> 5, lane = tid & 31;
  const int h = lane >> 4, ln = lane & 15;
  const int b = blockIdx.x / NCH3, ch = blockIdx.x % NCH3;
  const int wave_u = __builtin_amdgcn_readfirstlane(wave);

  const float* xbase =
      xg + (size_t)b * CC * NN * TT + (size_t)(ch * NODES3) * TT;
  if (wave_u == 0)
    tdm_load_2d_f32(xbase, lds_off_of(sXf0), TT, CC,
                    (unsigned long long)NN * TT);

  for (int i = tid; i < 256 * 128; i += 256) sW3[i] = (__bf16)W3[i];
  for (int i = tid; i < 64 * 256; i += 256) sW4[i] = (__bf16)W4[i];
  for (int i = tid; i < 4096; i += 256) sPool[i] = pooled[b * 4096 + i];
  for (int i = tid; i < 256; i += 256) sb3[i] = b3[i];
  if (tid < 64) {
    sb4[tid] = b4[tid];
    const float sc = gamma[tid] * rsqrtf(rvar[tid] + BN_EPS);
    sScale[tid] = sc;
    sShift[tid] = beta[tid] - rmean[tid] * sc;
  }
  __syncthreads();
  // upper half of cat (= pooled, broadcast over n) is loop-invariant
  for (int i = tid; i < 4096; i += 256) {
    const int c = i >> 6, t = i & 63;
    sCat[t * 128 + 64 + c] = (__bf16)sPool[c * 64 + t];
  }

  for (int i = 0; i < NODES3; ++i) {
    const int n = ch * NODES3 + i;
    float* cur = (i & 1) ? sXf1 : sXf0;
    float* nxt = (i & 1) ? sXf0 : sXf1;
    if (wave_u == 0) __builtin_amdgcn_s_wait_tensorcnt(0);
    __syncthreads();  // tile i visible; prev iter GEMM4 reads of nxt done
    if (wave_u == 0 && i + 1 < NODES3)
      tdm_load_2d_f32(xbase + (size_t)(i + 1) * TT, lds_off_of(nxt), TT, CC,
                      (unsigned long long)NN * TT);

    // build varying lower half of cat: bf16(x - pooled), [t][k] layout
    for (int idx = tid; idx < 4096; idx += 256) {
      const int c = idx >> 6, t = idx & 63;
      sCat[t * 128 + c] = (__bf16)(cur[c * 64 + t] - sPool[c * 64 + t]);
    }
    __syncthreads();

    // GEMM3: h = gelu(W3 cat + b3) : M=256, N=64, K=128 ; 8 frags/wave
#pragma unroll
    for (int j = 0; j < 8; ++j) {
      const int slot = wave * 8 + j, mt = slot >> 2, tt = slot & 3;
      fx8 acc = {};
#pragma unroll
      for (int k0 = 0; k0 < 128; k0 += 32)
        acc = wmma_bf16(ldA(sW3, 128, lane, mt * 16, k0),
                        ldB(sCat, 128, lane, tt * 16, k0), acc);
      const int pos = tt * 16 + ln;
      bfx8 ho;
#pragma unroll
      for (int v = 0; v < 8; ++v)
        ho[v] = (__bf16)gelu_exact(acc[v] + sb3[mt * 16 + h * 8 + v]);
      *(bfx8*)(sH + pos * 256 + mt * 16 + h * 8) = ho;
    }
    __syncthreads();

    // GEMM4: y = BN(W4 h + b4 + x) : M=64, N=64, K=256 ; 2 frags/wave
#pragma unroll
    for (int j = 0; j < 2; ++j) {
      const int slot = wave * 2 + j, mt = slot >> 2, tt = slot & 3;
      fx8 acc = {};
#pragma unroll
      for (int k0 = 0; k0 < 256; k0 += 32)
        acc = wmma_bf16(ldA(sW4, 256, lane, mt * 16, k0),
                        ldB(sH, 256, lane, tt * 16, k0), acc);
      const int t = tt * 16 + ln;
#pragma unroll
      for (int v = 0; v < 8; ++v) {
        const int c = mt * 16 + h * 8 + v;
        const float hv = acc[v] + sb4[c] + cur[c * 64 + t];
        out[((size_t)(b * CC + c) * NN + n) * TT + t] =
            hv * sScale[c] + sShift[c];
      }
    }
  }
}

// ---------------------------------------------------------------------------
extern "C" void kernel_launch(void* const* d_in, const int* in_sizes, int n_in,
                              void* d_out, int out_size, void* d_ws,
                              size_t ws_size, hipStream_t stream) {
  const float* x = (const float*)d_in[0];
  const float* W1 = (const float*)d_in[1];
  const float* b1 = (const float*)d_in[2];
  const float* W2 = (const float*)d_in[3];
  const float* b2 = (const float*)d_in[4];
  const float* W5 = (const float*)d_in[5];
  const float* b5 = (const float*)d_in[6];
  const float* W3 = (const float*)d_in[7];
  const float* b3 = (const float*)d_in[8];
  const float* W4 = (const float*)d_in[9];
  const float* b4 = (const float*)d_in[10];
  const float* gamma = (const float*)d_in[11];
  const float* beta = (const float*)d_in[12];
  const float* rmean = (const float*)d_in[13];
  const float* rvar = (const float*)d_in[14];
  float* out = (float*)d_out;

  float* part = (float*)d_ws;                              // B*NCH1*3*C*T f32
  float* pooled = part + (size_t)BB * NCH1 * 3 * CC * TT;  // + B*C*T f32

  // k1 ~74 KB, k3 ~194 KB dynamic LDS (WGP allows up to 320 KB per workgroup)
  (void)hipFuncSetAttribute((const void*)k1_pool_partials,
                            hipFuncAttributeMaxDynamicSharedMemorySize,
                            96 * 1024);
  (void)hipFuncSetAttribute((const void*)k3_mlp_bn,
                            hipFuncAttributeMaxDynamicSharedMemorySize,
                            224 * 1024);

  const size_t sm1 = 5 * 4096 * sizeof(__bf16) +
                     (3 * 64 + 2 * 4096) * sizeof(float);
  k1_pool_partials<<<dim3(BB * NCH1), dim3(256), sm1, stream>>>(
      x, W1, b1, W2, b2, W5, b5, part);

  k2_pool_reduce<<<dim3((BB * CC * TT + 255) / 256), dim3(256), 0, stream>>>(
      part, pooled);

  const size_t sm3 = (size_t)(256 * 128 + 64 * 256 + 64 * 128 + 64 * 256) *
                         sizeof(__bf16) +
                     (size_t)(3 * 4096 + 256 + 64 + 64 + 64) * sizeof(float);
  k3_mlp_bn<<<dim3(BB * NCH3), dim3(256), sm3, stream>>>(
      x, pooled, W3, b3, W4, b4, gamma, beta, rmean, rvar, out);
}
